// MultiHeadAttention_1477468749913
// MI455X (gfx1250) — compile-verified, matrix operand order checked by reading, not run on hardware
//
#include <hip/hip_runtime.h>
#include <hip/hip_bf16.h>

typedef __attribute__((ext_vector_type(16))) _Float16 v16h;
typedef __attribute__((ext_vector_type(8)))  _Float16 v8h;
typedef __attribute__((ext_vector_type(8)))  float    v8f;
typedef __attribute__((ext_vector_type(4)))  int      v4i;

constexpr float SCALE = 0.03125f;   // 1/sqrt(1024)

// ---------------------------------------------------------------------------
// CDNA5 async global->LDS staging (double-buffer path), with sync fallback.
// Builtin signature (probed via diagnostic): (int4 global*, int4 shared*, Ii, Ii)
// ---------------------------------------------------------------------------
#if defined(__has_builtin)
# if __has_builtin(__builtin_amdgcn_global_load_async_to_lds_b128)
#  define HAS_ASYNC 1
# endif
#endif
#ifndef HAS_ASYNC
# define HAS_ASYNC 0
#endif

#if HAS_ASYNC
typedef __attribute__((address_space(1))) v4i* gv4i_p;
typedef __attribute__((address_space(3))) v4i* lv4i_p;
#endif

__device__ __forceinline__ void stage_b128(const _Float16* g, _Float16* l) {
#if HAS_ASYNC
  __builtin_amdgcn_global_load_async_to_lds_b128(
      (gv4i_p)(unsigned long long)(const void*)g,
      (lv4i_p)(unsigned int)(unsigned long long)(const void*)l,
      0, 0);
#else
  *(v8h*)l = *(const v8h*)g;
#endif
}

// Wait until at most `n` async ops outstanding (16: current tile done, next
// tile still in flight; 0: everything done).
__device__ __forceinline__ void stage_wait16() {
#if HAS_ASYNC
  asm volatile("s_wait_asynccnt 0x10" ::: "memory");
#endif
}
__device__ __forceinline__ void stage_wait0() {
#if HAS_ASYNC
  asm volatile("s_wait_asynccnt 0x0" ::: "memory");
#endif
}

// ---------------------------------------------------------------------------
// WMMA helper: D = A(16x32 f16) * B(32x16 f16) + C(16x16 f32)
// ---------------------------------------------------------------------------
__device__ __forceinline__ v8f wmma_f16(v16h a, v16h b, v8f c) {
  return __builtin_amdgcn_wmma_f32_16x16x32_f16(
      false, a, false, b, (short)0, c, false, false);
}

// ---------------------------------------------------------------------------
// 16x32 f16 fragment from row-major source (ld elements). ISA 16-bit layout:
//   lanes 0-15 : row=lane,    elems 0..7 -> K 0..7,  8..15 -> K 16..23
//   lanes 16-31: row=lane-16, elems 0..7 -> K 8..15, 8..15 -> K 24..31
// => two contiguous 16B loads per lane. Works for B fragments when the B
// matrix is stored transposed (source row = output column).
// ---------------------------------------------------------------------------
__device__ __forceinline__ v16h load_frag(const _Float16* base, int ld, int lane) {
  const int r   = lane & 15;
  const int sel = (lane >> 4) << 3;
  const _Float16* p = base + r * ld + sel;
  v8h lo = *(const v8h*)(p);
  v8h hi = *(const v8h*)(p + 16);
  v16h a;
#pragma unroll
  for (int i = 0; i < 8; ++i) { a[i] = lo[i]; a[i + 8] = hi[i]; }
  return a;
}

// ---------------------------------------------------------------------------
// Kernel 1: f32 -> f16 cast (x)
// ---------------------------------------------------------------------------
__global__ void cast_f16_kernel(const float* __restrict__ in,
                                _Float16* __restrict__ out, int n) {
  int i = blockIdx.x * blockDim.x + threadIdx.x;
  if (i < n) out[i] = (_Float16)in[i];
}

// ---------------------------------------------------------------------------
// Kernel 2: 1024x1024 f32 -> f16 transpose (Wt[o][i] = W[i][o])
// ---------------------------------------------------------------------------
__global__ void transpose_cast_kernel(const float* __restrict__ in,
                                      _Float16* __restrict__ out) {
  __shared__ float tile[32][33];
  const int tx = threadIdx.x, ty = threadIdx.y;
  const int i0 = blockIdx.y * 32;
  const int o0 = blockIdx.x * 32;
  tile[ty][tx] = in[(size_t)(i0 + ty) * 1024 + (o0 + tx)];
  __syncthreads();
  out[(size_t)(o0 + ty) * 1024 + (i0 + tx)] = (_Float16)tile[tx][ty];
}

// ---------------------------------------------------------------------------
// Kernel 3: QKV projection. One wave -> 32(M) x 64(N) tile of one of Q/K/V.
// grid = (128, 16, 3), block = 32. Q,K: [b,h,n,dh]; V transposed [b,h,dh,n].
// ---------------------------------------------------------------------------
__global__ __launch_bounds__(32)
void qkv_proj_kernel(const _Float16* __restrict__ xh,
                     const _Float16* __restrict__ wqt,
                     const _Float16* __restrict__ wkt,
                     const _Float16* __restrict__ wvt,
                     _Float16* __restrict__ Qd,
                     _Float16* __restrict__ Kd,
                     _Float16* __restrict__ Vt) {
  const int lane  = threadIdx.x;
  const int mbase = blockIdx.x * 32;
  const int nbase = blockIdx.y * 64;
  const int w     = blockIdx.z;
  const _Float16* W = (w == 0) ? wqt : (w == 1) ? wkt : wvt;

  v8f acc[2][4];
#pragma unroll
  for (int t = 0; t < 2; ++t)
#pragma unroll
    for (int j = 0; j < 4; ++j)
#pragma unroll
      for (int e = 0; e < 8; ++e) acc[t][j][e] = 0.0f;

  const _Float16* a0p = xh + (size_t)mbase * 1024;
  const _Float16* a1p = xh + (size_t)(mbase + 16) * 1024;
  const _Float16* w0p = W + (size_t)(nbase + 0)  * 1024;
  const _Float16* w1p = W + (size_t)(nbase + 16) * 1024;
  const _Float16* w2p = W + (size_t)(nbase + 32) * 1024;
  const _Float16* w3p = W + (size_t)(nbase + 48) * 1024;

  for (int k = 0; k < 1024; k += 32) {
    __builtin_prefetch(a0p + k + 256, 0, 1);
    v16h a0 = load_frag(a0p + k, 1024, lane);
    v16h a1 = load_frag(a1p + k, 1024, lane);
    v16h b0 = load_frag(w0p + k, 1024, lane);
    v16h b1 = load_frag(w1p + k, 1024, lane);
    v16h b2 = load_frag(w2p + k, 1024, lane);
    v16h b3 = load_frag(w3p + k, 1024, lane);
    acc[0][0] = wmma_f16(a0, b0, acc[0][0]);
    acc[1][0] = wmma_f16(a1, b0, acc[1][0]);
    acc[0][1] = wmma_f16(a0, b1, acc[0][1]);
    acc[1][1] = wmma_f16(a1, b1, acc[1][1]);
    acc[0][2] = wmma_f16(a0, b2, acc[0][2]);
    acc[1][2] = wmma_f16(a1, b2, acc[1][2]);
    acc[0][3] = wmma_f16(a0, b3, acc[0][3]);
    acc[1][3] = wmma_f16(a1, b3, acc[1][3]);
  }

  const int half = lane >> 4;
  const int cn   = lane & 15;
  if (w < 2) {
    _Float16* dst = (w == 0) ? Qd : Kd;
#pragma unroll
    for (int t = 0; t < 2; ++t)
#pragma unroll
      for (int j = 0; j < 4; ++j) {
        const int col = nbase + 16 * j + cn;
        const int h = col >> 6, dh = col & 63;
#pragma unroll
        for (int r = 0; r < 8; ++r) {
          const int tok = mbase + 16 * t + 8 * half + r;
          const int b = tok >> 11, n = tok & 2047;
          dst[(((size_t)(b * 16 + h)) * 2048 + n) * 64 + dh] = (_Float16)acc[t][j][r];
        }
      }
  } else {
#pragma unroll
    for (int t = 0; t < 2; ++t)
#pragma unroll
      for (int j = 0; j < 4; ++j) {
        const int col = nbase + 16 * j + cn;
        const int h = col >> 6, dh = col & 63;
        const int tok0 = mbase + 16 * t + 8 * half;
        const int b = tok0 >> 11, n0 = tok0 & 2047;
        v8h pk;
#pragma unroll
        for (int r = 0; r < 8; ++r) pk[r] = (_Float16)acc[t][j][r];
        *(v8h*)(Vt + (((size_t)(b * 16 + h)) * 64 + dh) * 2048 + n0) = pk;
      }
  }
}

// ---------------------------------------------------------------------------
// Kernel 4: causal flash attention. One wave per (b*h, 32-query tile).
// grid = (N/32=64, B*H=32), block = 32.
// K (32x64) and V^T (64x32) tiles double-buffered in LDS via async loads.
// ---------------------------------------------------------------------------
__global__ __launch_bounds__(32)
void attn_kernel(const _Float16* __restrict__ Qd,
                 const _Float16* __restrict__ Kd,
                 const _Float16* __restrict__ Vt,
                 _Float16* __restrict__ ctx) {
  __shared__ alignas(16) _Float16 Ks[2][32 * 64];  // [key][dh]   4 KB each
  __shared__ alignas(16) _Float16 Vs[2][64 * 32];  // [dh][key]   4 KB each
  __shared__ alignas(16) _Float16 Pl[2][16 * 32];  // P tiles, C->A relayout

  const int lane  = threadIdx.x;
  const int qbase = blockIdx.x * 32;
  const int bh    = blockIdx.y;
  const int half  = lane >> 4, cn = lane & 15;

  const _Float16* Qb = Qd + (size_t)bh * (2048 * 64);
  const _Float16* Kb = Kd + (size_t)bh * (2048 * 64);
  const _Float16* Vb = Vt + (size_t)bh * (64 * 2048);

  v16h aQ[2][2];
#pragma unroll
  for (int t = 0; t < 2; ++t) {
    aQ[t][0] = load_frag(Qb + (size_t)(qbase + 16 * t) * 64 + 0,  64, lane);
    aQ[t][1] = load_frag(Qb + (size_t)(qbase + 16 * t) * 64 + 32, 64, lane);
  }

  v8f acc[2][4];
#pragma unroll
  for (int t = 0; t < 2; ++t)
#pragma unroll
    for (int j = 0; j < 4; ++j)
#pragma unroll
      for (int e = 0; e < 8; ++e) acc[t][j][e] = 0.0f;

  float m_i[2][8], l_i[2][8];
#pragma unroll
  for (int t = 0; t < 2; ++t)
#pragma unroll
    for (int r = 0; r < 8; ++r) { m_i[t][r] = -__builtin_inff(); l_i[t][r] = 0.0f; }

  const int vrow = lane >> 2;            // 0..7
  const int vcol = (lane & 3) * 8;       // 0,8,16,24 halves
  const int nkt  = (qbase + 31) / 32 + 1;

  // ---- prologue: stage tile 0 (16 async b128 issues)
  {
    const _Float16* kg = Kb;             // kb = 0, contiguous 4 KB
#pragma unroll
    for (int i = 0; i < 8; ++i)
      stage_b128(kg + i * 256 + lane * 8, &Ks[0][i * 256 + lane * 8]);
#pragma unroll
    for (int i = 0; i < 8; ++i)
      stage_b128(Vb + (size_t)(i * 8 + vrow) * 2048 + vcol,
                 &Vs[0][(i * 8 + vrow) * 32 + vcol]);
  }

  for (int kt = 0; kt < nkt; ++kt) {
    const int kb  = kt * 32;
    const int buf = kt & 1;

    // ---- stage next tile while current is consumed (double buffer)
    if (kt + 1 < nkt) {
      const int kb2 = kb + 32;
      const _Float16* kg = Kb + (size_t)kb2 * 64;
#pragma unroll
      for (int i = 0; i < 8; ++i)
        stage_b128(kg + i * 256 + lane * 8, &Ks[buf ^ 1][i * 256 + lane * 8]);
#pragma unroll
      for (int i = 0; i < 8; ++i)
        stage_b128(Vb + (size_t)(i * 8 + vrow) * 2048 + kb2 + vcol,
                   &Vs[buf ^ 1][(i * 8 + vrow) * 32 + vcol]);
      stage_wait16();                    // current tile's 16 issues complete
    } else {
      stage_wait0();
    }
#if !HAS_ASYNC
    __syncthreads();
#endif

    // ---- scores: S[2][16x32] = Q * K^T  (K frags straight from LDS)
    v16h kf0 = load_frag(&Ks[buf][0 * 64 + 0],  64, lane);
    v16h kf1 = load_frag(&Ks[buf][0 * 64 + 32], 64, lane);
    v16h kf2 = load_frag(&Ks[buf][16 * 64 + 0], 64, lane);
    v16h kf3 = load_frag(&Ks[buf][16 * 64 + 32], 64, lane);
    v8f s0[2], s1[2];
#pragma unroll
    for (int t = 0; t < 2; ++t) {
#pragma unroll
      for (int e = 0; e < 8; ++e) { s0[t][e] = 0.0f; s1[t][e] = 0.0f; }
      s0[t] = wmma_f16(aQ[t][0], kf0, s0[t]);
      s0[t] = wmma_f16(aQ[t][1], kf1, s0[t]);
      s1[t] = wmma_f16(aQ[t][0], kf2, s1[t]);
      s1[t] = wmma_f16(aQ[t][1], kf3, s1[t]);
    }

    // ---- scale + causal mask + online softmax per q-block
#pragma unroll
    for (int t = 0; t < 2; ++t) {
      const int qmin = qbase + 16 * t;
      const bool need_mask = (kb + 31 > qmin);
      float p0[8], p1[8];
#pragma unroll
      for (int r = 0; r < 8; ++r) {
        const int qg = qmin + 8 * half + r;
        float v0 = s0[t][r] * SCALE;
        float v1 = s1[t][r] * SCALE;
        if (need_mask) {
          if (kb + cn > qg)      v0 = -__builtin_inff();
          if (kb + 16 + cn > qg) v1 = -__builtin_inff();
        }
        p0[r] = v0; p1[r] = v1;
      }
#pragma unroll
      for (int r = 0; r < 8; ++r) {
        float mx = fmaxf(p0[r], p1[r]);
#pragma unroll
        for (int m = 1; m <= 8; m <<= 1) mx = fmaxf(mx, __shfl_xor(mx, m, 32));
        const float mnew  = fmaxf(m_i[t][r], mx);
        const float alpha = __expf(m_i[t][r] - mnew);
        const float e0 = __expf(p0[r] - mnew);
        const float e1 = __expf(p1[r] - mnew);
        float rs = e0 + e1;
#pragma unroll
        for (int m = 1; m <= 8; m <<= 1) rs += __shfl_xor(rs, m, 32);
        l_i[t][r] = l_i[t][r] * alpha + rs;
        m_i[t][r] = mnew;
        p0[r] = e0; p1[r] = e1;
#pragma unroll
        for (int j = 0; j < 4; ++j) acc[t][j][r] *= alpha;
      }
      // P: C-layout -> LDS (row-major 16x32) for A-fragment reload
#pragma unroll
      for (int r = 0; r < 8; ++r) {
        Pl[t][(8 * half + r) * 32 + cn]      = (_Float16)p0[r];
        Pl[t][(8 * half + r) * 32 + 16 + cn] = (_Float16)p1[r];
      }
    }
    __syncthreads();   // single-wave block: DS-order fence, barrier ~ free

    const v16h aP0 = load_frag(&Pl[0][0], 32, lane);
    const v16h aP1 = load_frag(&Pl[1][0], 32, lane);

    // ---- ctx += P * V  (V^T frags from LDS, shared across both q-blocks)
#pragma unroll
    for (int j = 0; j < 4; ++j) {
      v16h bV = load_frag(&Vs[buf][j * 16 * 32], 32, lane);
      acc[0][j] = wmma_f16(aP0, bV, acc[0][j]);
      acc[1][j] = wmma_f16(aP1, bV, acc[1][j]);
    }
    __syncthreads();
  }

  // ---- finalize: divide by row sums, store ctx [b, n, D] f16
  const int b = bh >> 4, h = bh & 15;
#pragma unroll
  for (int t = 0; t < 2; ++t) {
    float inv[8];
#pragma unroll
    for (int r = 0; r < 8; ++r) inv[r] = 1.0f / l_i[t][r];
#pragma unroll
    for (int j = 0; j < 4; ++j) {
      const int col = h * 64 + j * 16 + cn;
#pragma unroll
      for (int r = 0; r < 8; ++r) {
        const int n = qbase + 16 * t + 8 * half + r;
        ctx[((size_t)(b * 2048 + n)) * 1024 + col] = (_Float16)(acc[t][j][r] * inv[r]);
      }
    }
  }
}

// ---------------------------------------------------------------------------
// Kernel 5: output projection  out = ctx @ Wo + bo  (f32). 32x64 per wave.
// grid = (128, 16), block = 32.
// ---------------------------------------------------------------------------
__global__ __launch_bounds__(32)
void out_proj_kernel(const _Float16* __restrict__ ctx,
                     const _Float16* __restrict__ wot,
                     const float* __restrict__ bo,
                     float* __restrict__ out) {
  const int lane  = threadIdx.x;
  const int mbase = blockIdx.x * 32;
  const int nbase = blockIdx.y * 64;

  v8f acc[2][4];
#pragma unroll
  for (int t = 0; t < 2; ++t)
#pragma unroll
    for (int j = 0; j < 4; ++j)
#pragma unroll
      for (int e = 0; e < 8; ++e) acc[t][j][e] = 0.0f;

  const _Float16* a0p = ctx + (size_t)mbase * 1024;
  const _Float16* a1p = ctx + (size_t)(mbase + 16) * 1024;
  const _Float16* w0p = wot + (size_t)(nbase + 0)  * 1024;
  const _Float16* w1p = wot + (size_t)(nbase + 16) * 1024;
  const _Float16* w2p = wot + (size_t)(nbase + 32) * 1024;
  const _Float16* w3p = wot + (size_t)(nbase + 48) * 1024;

  for (int k = 0; k < 1024; k += 32) {
    __builtin_prefetch(a0p + k + 256, 0, 1);
    v16h a0 = load_frag(a0p + k, 1024, lane);
    v16h a1 = load_frag(a1p + k, 1024, lane);
    v16h b0 = load_frag(w0p + k, 1024, lane);
    v16h b1 = load_frag(w1p + k, 1024, lane);
    v16h b2 = load_frag(w2p + k, 1024, lane);
    v16h b3 = load_frag(w3p + k, 1024, lane);
    acc[0][0] = wmma_f16(a0, b0, acc[0][0]);
    acc[1][0] = wmma_f16(a1, b0, acc[1][0]);
    acc[0][1] = wmma_f16(a0, b1, acc[0][1]);
    acc[1][1] = wmma_f16(a1, b1, acc[1][1]);
    acc[0][2] = wmma_f16(a0, b2, acc[0][2]);
    acc[1][2] = wmma_f16(a1, b2, acc[1][2]);
    acc[0][3] = wmma_f16(a0, b3, acc[0][3]);
    acc[1][3] = wmma_f16(a1, b3, acc[1][3]);
  }

  const int half = lane >> 4;
  const int cn   = lane & 15;
#pragma unroll
  for (int t = 0; t < 2; ++t)
#pragma unroll
    for (int j = 0; j < 4; ++j) {
      const int col = nbase + 16 * j + cn;
      const float bias = bo[col];
#pragma unroll
      for (int r = 0; r < 8; ++r) {
        const int tok = mbase + 16 * t + 8 * half + r;
        out[(size_t)tok * 1024 + col] = acc[t][j][r] + bias;
      }
    }
}

// ---------------------------------------------------------------------------
// Host launcher
// ---------------------------------------------------------------------------
extern "C" void kernel_launch(void* const* d_in, const int* in_sizes, int n_in,
                              void* d_out, int out_size, void* d_ws, size_t ws_size,
                              hipStream_t stream) {
  (void)in_sizes; (void)n_in; (void)out_size; (void)ws_size;

  const float* x  = (const float*)d_in[0];
  const float* Wq = (const float*)d_in[1];
  const float* Wk = (const float*)d_in[2];
  const float* Wv = (const float*)d_in[3];
  const float* Wo = (const float*)d_in[4];
  const float* bo = (const float*)d_in[5];
  float* out = (float*)d_out;

  constexpr size_t MB = 1024ull * 1024ull;
  char* ws = (char*)d_ws;
  _Float16* xh  = (_Float16*)(ws + 0);        //  8 MB: x f16 [4096,1024]
  _Float16* wqt = (_Float16*)(ws + 8 * MB);   //  2 MB: Wq^T
  _Float16* wkt = (_Float16*)(ws + 10 * MB);  //  2 MB
  _Float16* wvt = (_Float16*)(ws + 12 * MB);  //  2 MB
  _Float16* wot = (_Float16*)(ws + 14 * MB);  //  2 MB
  _Float16* Qd  = (_Float16*)(ws + 16 * MB);  //  8 MB: Q [b,h,n,dh]
  _Float16* Kd  = (_Float16*)(ws + 24 * MB);  //  8 MB: K [b,h,n,dh]
  _Float16* Vt  = (_Float16*)(ws + 32 * MB);  //  8 MB: V^T [b,h,dh,n]
  _Float16* ctx = (_Float16*)(ws + 40 * MB);  //  8 MB: ctx [b,n,D]  (48 MB total)

  const int nx = 2 * 2048 * 1024;
  cast_f16_kernel<<<(nx + 255) / 256, 256, 0, stream>>>(x, xh, nx);

  dim3 tb(32, 32), tg(32, 32);
  transpose_cast_kernel<<<tg, tb, 0, stream>>>(Wq, wqt);
  transpose_cast_kernel<<<tg, tb, 0, stream>>>(Wk, wkt);
  transpose_cast_kernel<<<tg, tb, 0, stream>>>(Wv, wvt);
  transpose_cast_kernel<<<tg, tb, 0, stream>>>(Wo, wot);

  qkv_proj_kernel<<<dim3(128, 16, 3), 32, 0, stream>>>(xh, wqt, wkt, wvt, Qd, Kd, Vt);
  attn_kernel<<<dim3(64, 32), 32, 0, stream>>>(Qd, Kd, Vt, ctx);
  out_proj_kernel<<<dim3(128, 16), 32, 0, stream>>>(ctx, wot, bo, out);
}
